// GTN_10144712753894
// MI455X (gfx1250) — compile-verified, matrix operand order checked by reading, not run on hardware
//
#include <hip/hip_runtime.h>

typedef _Float16 v16h __attribute__((ext_vector_type(16)));
typedef _Float16 h8   __attribute__((ext_vector_type(8)));
typedef _Float16 h4   __attribute__((ext_vector_type(4)));
typedef float    v8f  __attribute__((ext_vector_type(8)));
typedef unsigned int u32x4 __attribute__((ext_vector_type(4)));
typedef int          i32x4 __attribute__((ext_vector_type(4)));
typedef int          i32x8 __attribute__((ext_vector_type(8)));

#define GB 64     // batch
#define GN 256    // set size
#define GD 128    // feature dim
#define GH 4      // heads
#define HB 256    // GH*GB problems
#define CLD 257   // padded LDS row stride for C (bank-conflict-free both ways)

// LDS layout for gtn_sinkhorn (bytes)
#define SMALL_FLOATS (256 * CLD + 8 * 256 + 8)          // C + 8 arrays + red
#define STAGE_OFF    (SMALL_FLOATS * 4)                 // 271392, 16B aligned
#define STAGE_LD     132                                 // halfs per staged row
#define STAGE_BYTES  (128 * STAGE_LD * 2)               // 33792
#define LDS_TOTAL    (STAGE_OFF + STAGE_BYTES)          // 305184 < 320KB

// ---------------------------------------------------------------------------
// Kernel 1: heads projection  h = (x * 1/sqrt(D)) @ W_dt   via WMMA f16->f32
// ---------------------------------------------------------------------------
__global__ __launch_bounds__(256) void gtn_heads(const float* __restrict__ x,
                                                 const float* __restrict__ Wdt,
                                                 float* __restrict__ hout) {
  const int lane  = threadIdx.x & 31;
  const int wid   = threadIdx.x >> 5;
  const int lhalf = lane >> 4;
  const int lmod  = lane & 15;
  const int tile  = blockIdx.x * 8 + wid;     // 0..32767
  const int tm    = tile >> 5;                // 0..1023 (row tiles)
  const int tc    = tile & 31;                // 0..31   (col tiles)
  const float escale = 0.08838834764831845f;  // 1/sqrt(128)

  const int arow = tm * 16 + lmod;
  const int col  = tc * 16 + lmod;

  v8f acc = {};
  for (int kc = 0; kc < GD; kc += 32) {
    const int k0 = kc + (lhalf ? 8 : 0);
    v16h av, bv;
    const float* ap = x + (size_t)arow * GD + k0;
    #pragma unroll
    for (int i = 0; i < 8; ++i) {
      av[i]     = (_Float16)(ap[i]      * escale);   // K = k0..k0+7
      av[8 + i] = (_Float16)(ap[16 + i] * escale);   // K = k0+16..k0+23
    }
    const float* bp = Wdt + (size_t)k0 * 512 + col;
    #pragma unroll
    for (int i = 0; i < 8; ++i) {
      bv[i]     = (_Float16)bp[(size_t)i * 512];
      bv[8 + i] = (_Float16)bp[(size_t)(16 + i) * 512];
    }
    acc = __builtin_amdgcn_wmma_f32_16x16x32_f16(false, av, false, bv,
                                                 (short)0, acc, false, false);
  }

  const int hd = col >> 7, d = col & 127;
  #pragma unroll
  for (int r = 0; r < 8; ++r) {
    const int M = tm * 16 + r + 8 * lhalf;    // global row = b*N + n
    const int b = M >> 8, n = M & 255;
    hout[((size_t)(hd * GB + b) * GN + n) * GD + d] = acc[r];
  }
}

// ---------------------------------------------------------------------------
// Kernel 2: per-row stats + f16 conversion.  One wave per row (2*65536 rows).
// ---------------------------------------------------------------------------
__global__ __launch_bounds__(256) void gtn_stats(const float* __restrict__ h1,
                                                 const float* __restrict__ h2,
                                                 const float* __restrict__ alpha,
                                                 _Float16* __restrict__ h1h,
                                                 _Float16* __restrict__ h2h,
                                                 float* __restrict__ nrm1,
                                                 float* __restrict__ nrm2,
                                                 float* __restrict__ delc,
                                                 float* __restrict__ insc) {
  const int lane = threadIdx.x & 31;
  const int wid  = threadIdx.x >> 5;
  const int row  = blockIdx.x * 8 + wid;      // 0..131071
  const int which = row >> 16;
  const int r = row & 65535;

  const float* src = which ? h2 : h1;
  _Float16*    dst = which ? h2h : h1h;

  const float4 v = *(const float4*)(src + (size_t)r * GD + lane * 4);
  const float4 a = *(const float4*)(alpha + lane * 4);
  float sn = v.x * v.x + v.y * v.y + v.z * v.z + v.w * v.w;
  const float dx = v.x - a.x, dy = v.y - a.y, dz = v.z - a.z, dw = v.w - a.w;
  float sd = dx * dx + dy * dy + dz * dz + dw * dw;

  h4 o;
  o[0] = (_Float16)v.x; o[1] = (_Float16)v.y;
  o[2] = (_Float16)v.z; o[3] = (_Float16)v.w;
  *(h4*)(dst + (size_t)r * GD + lane * 4) = o;

  for (int off = 16; off; off >>= 1) {
    sn += __shfl_xor(sn, off, 32);
    sd += __shfl_xor(sd, off, 32);
  }
  if (lane == 0) {
    if (which) { nrm2[r] = sn; insc[r] = sqrtf(sd + 1e-12f); }
    else       { nrm1[r] = sn; delc[r] = sqrtf(sd + 1e-12f); }
  }
}

// ---------------------------------------------------------------------------
// TDM helper: DMA one 128-row x 128-col f16 tile of h2 into LDS staging,
// with pad_enable inserting 8B per 256B row (LDS row stride = 132 halfs).
// ---------------------------------------------------------------------------
__device__ __forceinline__ void tdm_stage_h2(const _Float16* gsrc) {
  u32x4 g0 = {0, 0, 0, 0};
  i32x8 g1 = {0, 0, 0, 0, 0, 0, 0, 0};
  i32x4 g2 = {0, 0, 0, 0};
  i32x4 g3 = {0, 0, 0, 0};
  const unsigned long long ga = (unsigned long long)(uintptr_t)gsrc;
  g0[0] = 1u;                                    // count=1 (valid user D#)
  g0[1] = (unsigned)STAGE_OFF;                   // lds_addr (bytes)
  g0[2] = (unsigned)(ga & 0xffffffffu);          // global_addr[31:0]
  g0[3] = (unsigned)((ga >> 32) & 0x01ffffffu)   // global_addr[56:32]
        | (2u << 30);                            // type = 2 ("image")
  // data_size=1(2B) | pad_enable | pad_interval=5(64 DW=256B) | pad_amount=1(2 DW=8B)
  g1[0] = (int)((1u << 16) | (1u << 20) | (5u << 22) | (1u << 25));
  g1[1] = (int)(128u << 16);                     // tensor_dim0 = 128 (low16)
  g1[2] = (int)(256u << 16);                     // tensor_dim1 = 256 (low16)
  g1[3] = (int)(128u << 16);                     // tile_dim0 = 128
  g1[4] = 128;                                   // tile_dim1 = 128 rows
  g1[5] = 128;                                   // tensor_dim0_stride = 128
#if __has_include(<hip/amd_detail/amd_gfx1250_TDM.h>)
  i32x8 g4 = {0, 0, 0, 0, 0, 0, 0, 0};
  __builtin_amdgcn_tensor_load_to_lds(g0, g1, g2, g3, g4, 0);
#else
  __builtin_amdgcn_tensor_load_to_lds(g0, g1, g2, g3, 0);
#endif
  __builtin_amdgcn_s_wait_tensorcnt(0);
}

// ---------------------------------------------------------------------------
// Kernel 3: one workgroup per (h,b) problem. TDM stages h2 (f16) into LDS in
// two 128-row halves; WMMA builds C (256x256, 257-padded) in LDS; 50 Sinkhorn
// iterations run entirely from LDS; final transport-cost contraction.
// ---------------------------------------------------------------------------
__global__ __launch_bounds__(256) void gtn_sinkhorn(
    const _Float16* __restrict__ h1h, const _Float16* __restrict__ h2h,
    const float* __restrict__ g_nrm1, const float* __restrict__ g_nrm2,
    const float* __restrict__ g_delc, const float* __restrict__ g_insc,
    const int* __restrict__ n1, const int* __restrict__ n2,
    float* __restrict__ g_dist) {
  extern __shared__ float smem[];
  float* sC   = smem;                 // 256*257
  float* sN1  = smem + 256 * CLD;
  float* sN2  = sN1 + 256;
  float* sDel = sN2 + 256;
  float* sIns = sDel + 256;
  float* sU1  = sIns + 256;
  float* sU2  = sU1 + 256;
  float* sV1  = sU2 + 256;
  float* sV2  = sV1 + 256;
  float* sRed = sV2 + 256;            // 8
  _Float16* sStage = (_Float16*)((char*)smem + STAGE_OFF);

  const int p = blockIdx.x;           // problem = hd*B + b
  const int t = threadIdx.x;
  const int lane = t & 31, wid = t >> 5;
  const int b = p & (GB - 1);
  const int n1r = n1[b], n2r = n2[b];
  const int nm = n1r < n2r ? n1r : n2r;
  const float eps  = 0.1f / logf((float)nm + 1.0f);
  const float k2   = 1.4426950408889634f / eps;   // log2(e)/eps
  const float cfac = eps * 0.6931471805599453f;   // eps*ln(2)

  sN1[t]  = g_nrm1[p * GN + t];
  sN2[t]  = g_nrm2[p * GN + t];
  sDel[t] = g_delc[p * GN + t];
  sIns[t] = g_insc[p * GN + t];
  sU1[t] = 0.f; sU2[t] = 0.f; sV1[t] = 0.f; sV2[t] = 0.f;
  __syncthreads();

  // ---- build C = sqrt(max(|h1|^2 + |h2|^2 - 2 h1.h2, 0) + 1e-12)
  const _Float16* A  = h1h + (size_t)p * (GN * GD);
  const _Float16* Bm = h2h + (size_t)p * (GN * GD);
  const int lhalf = lane >> 4, lmod = lane & 15;

  for (int half = 0; half < 2; ++half) {
    if (wid == 0) {                         // TDM: stage 128 rows of h2 -> LDS
      tdm_stage_h2(Bm + (size_t)half * 128 * GD);
    }
    __syncthreads();                        // staged data visible to all waves

    for (int tt = wid; tt < 128; tt += 8) { // 16 row-tiles x 8 col-tiles
      const int tm  = tt >> 3;
      const int tnl = tt & 7;
      const int tn  = half * 8 + tnl;
      const int arow  = tm * 16 + lmod;
      const int browL = tnl * 16 + lmod;    // row within staged half
      v8f acc = {};
      for (int kc = 0; kc < GD; kc += 32) {
        const int k0 = kc + (lhalf ? 8 : 0);
        h8 a0 = *(const h8*)(A + (size_t)arow * GD + k0);
        h8 a1 = *(const h8*)(A + (size_t)arow * GD + k0 + 16);
        h8 b0 = *(const h8*)(sStage + browL * STAGE_LD + k0);
        h8 b1 = *(const h8*)(sStage + browL * STAGE_LD + k0 + 16);
        v16h av = __builtin_shufflevector(a0, a1, 0,1,2,3,4,5,6,7,8,9,10,11,12,13,14,15);
        v16h bv = __builtin_shufflevector(b0, b1, 0,1,2,3,4,5,6,7,8,9,10,11,12,13,14,15);
        acc = __builtin_amdgcn_wmma_f32_16x16x32_f16(false, av, false, bv,
                                                     (short)0, acc, false, false);
      }
      #pragma unroll
      for (int r = 0; r < 8; ++r) {
        const int M  = tm * 16 + r + 8 * lhalf;
        const int Nc = tn * 16 + lmod;
        const float sq = sN1[M] + sN2[Nc] - 2.0f * acc[r];
        sC[M * CLD + Nc] = sqrtf(fmaxf(sq, 0.0f) + 1e-12f);
      }
    }
    __syncthreads();                        // protect staging re-use next half
  }

  // ---- 50 Sinkhorn iterations, all in LDS
  for (int it = 0; it < 50; ++it) {
    float u1n = 0.f, u2n = 0.f;
    if (t < n1r) {
      const float* crow = sC + t * CLD;
      const float tail = (sV2[t] - sDel[t]) * k2;
      float bm = tail;
      for (int m = 0; m < n2r; ++m) bm = fmaxf(bm, (sV1[m] - crow[m]) * k2);
      float s = exp2f(tail - bm);
      for (int m = 0; m < n2r; ++m) s += exp2f((sV1[m] - crow[m]) * k2 - bm);
      u1n = -cfac * (bm + log2f(s));
    }
    {
      float bm2 = -3.0e38f;
      for (int j = 0; j < n1r; ++j) bm2 = fmaxf(bm2, sV2[j] * k2);
      float s2 = 0.f;
      for (int j = 0; j < n1r; ++j) s2 += exp2f(sV2[j] * k2 - bm2);
      const float L2v2 = bm2 + log2f(s2);
      if (t < n2r) {
        const float a1 = (sV1[t] - sIns[t]) * k2;
        const float mx = fmaxf(a1, L2v2);
        u2n = -cfac * (mx + log2f(exp2f(a1 - mx) + exp2f(L2v2 - mx)));
      }
    }
    sU1[t] = u1n; sU2[t] = u2n;
    __syncthreads();

    float v1n = 0.f, v2n = 0.f;
    if (t < n2r) {
      const float tail = (sU2[t] - sIns[t]) * k2;
      float bm = tail;
      for (int j = 0; j < n1r; ++j) bm = fmaxf(bm, (sU1[j] - sC[j * CLD + t]) * k2);
      float s = exp2f(tail - bm);
      for (int j = 0; j < n1r; ++j) s += exp2f((sU1[j] - sC[j * CLD + t]) * k2 - bm);
      v1n = -cfac * (bm + log2f(s));
    }
    {
      float bm2 = -3.0e38f;
      for (int m = 0; m < n2r; ++m) bm2 = fmaxf(bm2, sU2[m] * k2);
      float s2 = 0.f;
      for (int m = 0; m < n2r; ++m) s2 += exp2f(sU2[m] * k2 - bm2);
      const float L2u2 = bm2 + log2f(s2);
      if (t < n1r) {
        const float a1 = (sU1[t] - sDel[t]) * k2;
        const float mx = fmaxf(a1, L2u2);
        v2n = -cfac * (mx + log2f(exp2f(a1 - mx) + exp2f(L2u2 - mx)));
      }
    }
    sV1[t] = v1n; sV2[t] = v2n;
    __syncthreads();
  }

  // ---- final: dist = sum T1*C + sum Td*delc + sum Ti*insc
  float part = 0.f;
  if (t < n1r) {
    const float* crow = sC + t * CLD;
    const float u1t = sU1[t];
    for (int m = 0; m < n2r; ++m) {
      const float c = crow[m];
      part += exp2f((u1t + sV1[m] - c) * k2) * c;
    }
    part += exp2f((u1t + sV2[t] - sDel[t]) * k2) * sDel[t];
  }
  if (t < n2r) part += exp2f((sU2[t] + sV1[t] - sIns[t]) * k2) * sIns[t];

  for (int off = 16; off; off >>= 1) part += __shfl_xor(part, off, 32);
  if (lane == 0) sRed[wid] = part;
  __syncthreads();
  if (t == 0) {
    float tot = 0.f;
    for (int w = 0; w < 8; ++w) tot += sRed[w];
    g_dist[p] = tot;
  }
}

// ---------------------------------------------------------------------------
// Kernel 4: MLP head  out[b] = leaky(d @ W1 + b1) @ W2 + b2
// ---------------------------------------------------------------------------
__global__ void gtn_mlp(const float* __restrict__ dist,
                        const float* __restrict__ W1, const float* __restrict__ b1,
                        const float* __restrict__ W2, const float* __restrict__ b2,
                        float* __restrict__ out) {
  const int b = threadIdx.x;
  if (b >= GB) return;
  float dd[GH];
  #pragma unroll
  for (int h = 0; h < GH; ++h) dd[h] = dist[h * GB + b];
  float acc = b2[0];
  for (int d = 0; d < GD; ++d) {
    float s = b1[d];
    #pragma unroll
    for (int h = 0; h < GH; ++h) s += dd[h] * W1[h * GD + d];
    s = s > 0.f ? s : 0.01f * s;
    acc += s * W2[d];
  }
  out[b] = acc;
}

// ---------------------------------------------------------------------------
extern "C" void kernel_launch(void* const* d_in, const int* in_sizes, int n_in,
                              void* d_out, int out_size, void* d_ws, size_t ws_size,
                              hipStream_t stream) {
  (void)in_sizes; (void)n_in; (void)out_size; (void)ws_size;
  const float* x1    = (const float*)d_in[0];
  const float* x2    = (const float*)d_in[1];
  const int*   n1    = (const int*)d_in[2];
  const int*   n2    = (const int*)d_in[3];
  const float* Wdt   = (const float*)d_in[4];
  const float* alpha = (const float*)d_in[5];
  const float* W1    = (const float*)d_in[6];
  const float* b1    = (const float*)d_in[7];
  const float* W2    = (const float*)d_in[8];
  const float* b2    = (const float*)d_in[9];
  float* out = (float*)d_out;

  const size_t HN = (size_t)HB * GN * GD;   // 8388608 elements
  float*    h1f  = (float*)d_ws;
  float*    h2f  = h1f + HN;
  _Float16* h1h  = (_Float16*)(h2f + HN);
  _Float16* h2h  = h1h + HN;
  float*    nrm1 = (float*)(h2h + HN);
  float*    nrm2 = nrm1 + HB * GN;
  float*    delc = nrm2 + HB * GN;
  float*    insc = delc + HB * GN;
  float*    dist = insc + HB * GN;

  gtn_heads<<<4096, 256, 0, stream>>>(x1, Wdt, h1f);
  gtn_heads<<<4096, 256, 0, stream>>>(x2, Wdt, h2f);
  gtn_stats<<<16384, 256, 0, stream>>>(h1f, h2f, alpha, h1h, h2h,
                                       nrm1, nrm2, delc, insc);

  hipFuncSetAttribute((const void*)gtn_sinkhorn,
                      hipFuncAttributeMaxDynamicSharedMemorySize, LDS_TOTAL);
  gtn_sinkhorn<<<HB, 256, (size_t)LDS_TOTAL, stream>>>(h1h, h2h, nrm1, nrm2,
                                                       delc, insc, n1, n2, dist);
  gtn_mlp<<<1, 64, 0, stream>>>(dist, W1, b1, W2, b2, out);
}
